// GradCacheColbertLoss_9139690406182
// MI455X (gfx1250) — matrix operand order repaired and checked by reading, not compiled, not run on hardware
//
#include <hip/hip_runtime.h>
#include <stdint.h>

// Problem constants (from reference): B=128, N=32, S=256, D=128, C=B=128
#define B_ 128
#define N_ 32
#define S_ 256
#define D_ 128
#define C_ 128

typedef __attribute__((ext_vector_type(16))) __bf16        v16bf;
typedef __attribute__((ext_vector_type(8)))  float         v8f;
typedef __attribute__((ext_vector_type(8)))  unsigned int  v8u;

// ---------------------------------------------------------------------------
// f32 -> bf16 (round-to-nearest-even), 4 elements / thread
// ---------------------------------------------------------------------------
__device__ __forceinline__ unsigned int f2bf(float f) {
  unsigned int u = __builtin_bit_cast(unsigned int, f);
  unsigned int r = u + 0x7fffu + ((u >> 16) & 1u);   // RNE (inputs are finite randn)
  return r >> 16;
}

__global__ void cvt_f32_to_bf16(const float4* __restrict__ in,
                                uint2* __restrict__ out, int n4) {
  int i = blockIdx.x * blockDim.x + threadIdx.x;
  if (i >= n4) return;
  float4 v = in[i];
  uint2 o;
  o.x = f2bf(v.x) | (f2bf(v.y) << 16);
  o.y = f2bf(v.z) | (f2bf(v.w) << 16);
  out[i] = o;
}

// ---------------------------------------------------------------------------
// Fragment loaders matching CDNA5 WMMA 16-bit operand VGPR layouts (wave32).
//
// A (16x32, MxK): lane = M (mod 16); elements 0..7  = K = kb..kb+7,
//                 elements 8..15 = K = kb+16..kb+23, kb = (lane>=16) ? 8 : 0.
// B (32x16, KxN): lane = N (mod 16); elements 0..7  = K = k0..k0+7,
//                 elements 8..15 = K = k0+16..k0+23, k0 = (lane>=16) ? 8 : 0.
// Both reduce to 16B-aligned b128 global loads of a row-major [*,128] matrix.
// ---------------------------------------------------------------------------
__device__ __forceinline__ v16bf load_frag_a(const unsigned short* __restrict__ p) {
  uint4 x = *(const uint4*)(p);        // K = kb .. kb+7
  uint4 y = *(const uint4*)(p + 16);   // K = kb+16 .. kb+23
  v8u t;
  t[0] = x.x; t[1] = x.y; t[2] = x.z; t[3] = x.w;
  t[4] = y.x; t[5] = y.y; t[6] = y.z; t[7] = y.w;
  return __builtin_bit_cast(v16bf, t);
}

__device__ __forceinline__ v16bf load_frag_b(const unsigned short* __restrict__ p) {
  uint4 x = *(const uint4*)(p);        // K = k0 .. k0+7
  uint4 y = *(const uint4*)(p + 16);   // K = k0+16 .. k0+23
  v8u t;
  t[0] = x.x; t[1] = x.y; t[2] = x.z; t[3] = x.w;
  t[4] = y.x; t[5] = y.y; t[6] = y.z; t[7] = y.w;
  return __builtin_bit_cast(v16bf, t);
}

// ---------------------------------------------------------------------------
// MaxSim kernel: one wave per (b,c). 8 waves / block share c (doc fragments
// are identical across the block's waves -> served by WGP$/L2).
// scores[b*C + c] = sum_n max_s  q[b,n,:] . d[c,s,:]
//
// Fully unrolled over the 16 s-tiles with software double-buffering of the
// doc (B) fragments: loads for tile st+1 are issued before the WMMAs of
// tile st, so the per-tile s_wait_loadcnt keeps 8 loads in flight.
// ---------------------------------------------------------------------------
__global__ __launch_bounds__(256) void maxsim_wmma(
    const unsigned short* __restrict__ Q,    // [B, N, D] bf16
    const unsigned short* __restrict__ Dm,   // [C, S, D] bf16
    float* __restrict__ scores)              // [B, C]
{
  const int wave = threadIdx.x >> 5;
  const int lane = threadIdx.x & 31;
  const int gw   = blockIdx.x * 8 + wave;    // 0 .. 16383
  const int b    = gw & (B_ - 1);            // 8 consecutive b per block
  const int c    = gw >> 7;                  // same c for whole block

  const int r   = lane & 15;
  const int kbA = (lane >> 4) << 3;          // 0 or 8   (A-fragment K base)
  const int k0B = (lane >> 4) << 3;          // 0 or 8   (B-fragment K base)

  const unsigned short* qrow0 = Q  + ((size_t)b * N_ + r) * D_ + kbA;       // n-tile 0
  const unsigned short* qrow1 = qrow0 + (size_t)16 * D_;                    // n-tile 1
  const unsigned short* drow  = Dm + ((size_t)c * S_ + r) * D_ + k0B;

  // Hoist all A fragments: 2 n-tiles x 4 k-steps (K=128 = 4 x 32)
  v16bf a0[4], a1[4];
#pragma unroll
  for (int kk = 0; kk < 4; ++kk) {
    a0[kk] = load_frag_a(qrow0 + kk * 32);
    a1[kk] = load_frag_a(qrow1 + kk * 32);
  }

  // Double-buffered B fragments
  v16bf cur[4], nxt[4];
#pragma unroll
  for (int kk = 0; kk < 4; ++kk) cur[kk] = load_frag_b(drow + kk * 32);

  v8f m0 = {}, m1 = {};
#pragma unroll
  for (int st = 0; st < 16; ++st) {
    // Prefetch next s-tile's fragments before consuming the current ones.
    if (st < 15) {
      const unsigned short* nb = drow + (size_t)(st + 1) * 16 * D_;
#pragma unroll
      for (int kk = 0; kk < 4; ++kk) nxt[kk] = load_frag_b(nb + kk * 32);
    }

    if (st == 0) {
      // First tile initializes the running max directly.
#pragma unroll
      for (int kk = 0; kk < 4; ++kk) {
        m0 = __builtin_amdgcn_wmma_f32_16x16x32_bf16(
            false, a0[kk], false, cur[kk], (short)0, m0, false, false);
        m1 = __builtin_amdgcn_wmma_f32_16x16x32_bf16(
            false, a1[kk], false, cur[kk], (short)0, m1, false, false);
      }
    } else {
      v8f acc0 = {}; v8f acc1 = {};
#pragma unroll
      for (int kk = 0; kk < 4; ++kk) {
        acc0 = __builtin_amdgcn_wmma_f32_16x16x32_bf16(
            false, a0[kk], false, cur[kk], (short)0, acc0, false, false);
        acc1 = __builtin_amdgcn_wmma_f32_16x16x32_bf16(
            false, a1[kk], false, cur[kk], (short)0, acc1, false, false);
      }
      // Elementwise running max over s-tiles (C layout: lane=n, VGPR=m)
#pragma unroll
      for (int i = 0; i < 8; ++i) {
        m0[i] = __builtin_fmaxf(m0[i], acc0[i]);
        m1[i] = __builtin_fmaxf(m1[i], acc1[i]);
      }
    }

    // Rotate buffers (pure register renaming after full unroll).
    if (st < 15) {
#pragma unroll
      for (int kk = 0; kk < 4; ++kk) cur[kk] = nxt[kk];
    }
  }

  // Max over n: 16-lane reduction within each half (xor masks < 16 keep the
  // two m-halves independent), then sum over m (VGPR index + both halves).
  float partial = 0.f;
#pragma unroll
  for (int i = 0; i < 8; ++i) {
    float v0 = m0[i], v1 = m1[i];
#pragma unroll
    for (int off = 8; off >= 1; off >>= 1) {
      v0 = __builtin_fmaxf(v0, __shfl_xor(v0, off, 32));
      v1 = __builtin_fmaxf(v1, __shfl_xor(v1, off, 32));
    }
    partial += v0 + v1;   // m = i and m = i+16 halves handled per lane-half
  }
  partial += __shfl_xor(partial, 16, 32);   // combine m<8(+16) with m>=8(+24)

  if (lane == 0) scores[(size_t)b * C_ + c] = partial;
}

// ---------------------------------------------------------------------------
// Cross-entropy with labels = arange(B): loss = mean_b( logsumexp(row_b) - row_b[b] )
// ---------------------------------------------------------------------------
__global__ void ce_loss(const float* __restrict__ scores, float* __restrict__ out) {
  __shared__ float red[B_];
  const int b = threadIdx.x;
  const float* row = scores + (size_t)b * C_;
  float mx = -__builtin_inff();
  for (int c = 0; c < C_; ++c) mx = __builtin_fmaxf(mx, row[c]);
  float se = 0.f;
  for (int c = 0; c < C_; ++c) se += __expf(row[c] - mx);
  red[b] = (mx + __logf(se)) - row[b];
  __syncthreads();
  for (int s = B_ / 2; s > 0; s >>= 1) {
    if (b < s) red[b] += red[b + s];
    __syncthreads();
  }
  if (b == 0) out[0] = red[0] / (float)B_;
}

// ---------------------------------------------------------------------------
extern "C" void kernel_launch(void* const* d_in, const int* in_sizes, int n_in,
                              void* d_out, int out_size, void* d_ws, size_t ws_size,
                              hipStream_t stream) {
  const float* q = (const float*)d_in[0];   // [128, 32, 128] f32
  const float* d = (const float*)d_in[1];   // [128, 256, 128] f32

  // Workspace layout: Q_bf16 (1 MB) | D_bf16 (8 MB) | scores (64 KB)
  unsigned short* Qbf = (unsigned short*)d_ws;
  unsigned short* Dbf = Qbf + (size_t)B_ * N_ * D_;
  float* scores = (float*)(Dbf + (size_t)C_ * S_ * D_);

  const int nq4 = (B_ * N_ * D_) / 4;   // 131072
  const int nd4 = (C_ * S_ * D_) / 4;   // 1048576
  cvt_f32_to_bf16<<<(nq4 + 255) / 256, 256, 0, stream>>>(
      (const float4*)q, (uint2*)Qbf, nq4);
  cvt_f32_to_bf16<<<(nd4 + 255) / 256, 256, 0, stream>>>(
      (const float4*)d, (uint2*)Dbf, nd4);

  // 16384 (b,c) waves, 8 waves / block
  maxsim_wmma<<<(B_ * C_) / 8, 256, 0, stream>>>(Qbf, Dbf, scores);

  ce_loss<<<1, B_, 0, stream>>>(scores, (float*)d_out);
}
